// SingleHeadAttention_53102975648107
// MI455X (gfx1250) — compile-verified, matrix operand order checked by reading, not run on hardware
//
#include <hip/hip_runtime.h>
#include <hip/hip_bf16.h>
#include <math.h>

// B=4, T=4096, E=1024, H=64 single-head causal attention, fp32 in/out.
// Strategy: bf16 WMMA (v_wmma_f32_16x16x32_bf16) for QKV GEMM, Q@K^T and P@V,
// fp32 accumulate + fp32 online softmax (flash-attention style).

typedef __attribute__((ext_vector_type(16))) __bf16 v16bf;
typedef __attribute__((ext_vector_type(8)))  float  v8f;

#define EDIM 1024
#define TDIM 4096
#define HDIM 64
#define BDIM 4

// ---------------------------------------------------------------------------
// Kernel 0: pack Wq/Wk/Wv (E x 64, fp32) into bf16 WMMA B-fragments.
// Fragment = 32(K) x 16(N) bf16 in ISA per-lane order; stored as
// frags[w][kt][nt][lane*16 + j], j = v16bf element index.
// B-layout (16-bit, 32x16): lanes 0-15: N=l, K = {0..7, 16..23};
//                           lanes 16-31: N=l-16, K = {8..15, 24..31}.
// ---------------------------------------------------------------------------
__global__ __launch_bounds__(32) void pack_weights_kernel(
    const float* __restrict__ Wq, const float* __restrict__ Wk,
    const float* __restrict__ Wv, __bf16* __restrict__ frags)
{
    const int fid = blockIdx.x;              // 0..383 = w*128 + kt*4 + nt
    const int w   = fid >> 7;                // 0..2
    const int kt  = (fid & 127) >> 2;        // 0..31
    const int nt  = fid & 3;                 // 0..3
    const float* W = (w == 0) ? Wq : ((w == 1) ? Wk : Wv);

    const int l  = threadIdx.x;              // 0..31
    const int n  = l & 15;
    const int hi = l >> 4;

    __bf16* dst = frags + (size_t)fid * 512 + l * 16;
#pragma unroll
    for (int j = 0; j < 16; ++j) {
        int K = (j < 8) ? (j + 8 * hi) : (j + 8 + 8 * hi);
        dst[j] = (__bf16)W[(kt * 32 + K) * HDIM + nt * 16 + n];
    }
}

// ---------------------------------------------------------------------------
// Kernel 1: QKV projection.  One wave computes a 16-row tile of q,k,v.
// A-layout (16-bit, 16x32): lanes 0-15: M=l, K={0..7,16..23};
//                           lanes 16-31: M=l-16, K={8..15,24..31}.
// q,k stored bf16 row-major (B*T,64); v stored bf16 TRANSPOSED (B,64,T).
// ---------------------------------------------------------------------------
__global__ __launch_bounds__(256) void qkv_kernel(
    const float*  __restrict__ x,            // (B*T, E)
    const float*  __restrict__ bq,
    const float*  __restrict__ bk,
    const float*  __restrict__ bv,
    const __bf16* __restrict__ frags,        // [3][32][4][512]
    __bf16* __restrict__ qb,                 // (B*T, 64)
    __bf16* __restrict__ kb,                 // (B*T, 64)
    __bf16* __restrict__ vT)                 // (B, 64, T)
{
    const int wave    = threadIdx.x >> 5;
    const int l       = threadIdx.x & 31;
    const int n       = l & 15;
    const int hi      = l >> 4;
    const int rowtile = blockIdx.x * 8 + wave;   // 0..1023
    const int row0    = rowtile * 16;

    v8f acc[3][4];
#pragma unroll
    for (int w = 0; w < 3; ++w)
#pragma unroll
        for (int t = 0; t < 4; ++t) acc[w][t] = (v8f)0.0f;

    const int   arow = row0 + n;             // A-fragment row for this lane
    const int   kofs = hi ? 8 : 0;
    const float* xr  = x + (size_t)arow * EDIM;

    for (int kt = 0; kt < 32; ++kt) {
        const float* xp = xr + kt * 32;
        v16bf a;
#pragma unroll
        for (int j = 0; j < 8; ++j) {
            a[j]     = (__bf16)xp[kofs + j];
            a[8 + j] = (__bf16)xp[16 + kofs + j];
        }
#pragma unroll
        for (int w = 0; w < 3; ++w) {
#pragma unroll
            for (int t = 0; t < 4; ++t) {
                const v16bf b =
                    *(const v16bf*)(frags + (((w * 32 + kt) * 4 + t) * 512) + l * 16);
                acc[w][t] = __builtin_amdgcn_wmma_f32_16x16x32_bf16(
                    false, a, false, b, (short)0, acc[w][t], false, false);
            }
        }
    }

    const int bidx = row0 >> 12;             // batch
    const int t0   = row0 & (TDIM - 1);      // token within batch
#pragma unroll
    for (int t = 0; t < 4; ++t) {
        const int   col = t * 16 + n;
        const float biq = bq[col];
        const float bik = bk[col];
        const float biv = bv[col];
#pragma unroll
        for (int r = 0; r < 8; ++r) {
            const int    m    = r + 8 * hi;
            const size_t grow = (size_t)(row0 + m);
            qb[grow * HDIM + col] = (__bf16)(acc[0][t][r] + biq);
            kb[grow * HDIM + col] = (__bf16)(acc[1][t][r] + bik);
            vT[((size_t)bidx * HDIM + col) * TDIM + (t0 + m)] =
                (__bf16)(acc[2][t][r] + biv);
        }
    }
}

// ---------------------------------------------------------------------------
// Kernel 2: causal flash attention.  One wave owns a 16-query tile and loops
// over 32-key blocks with online softmax.  C/D layout: lanes 0-15: N=l, row
// M = r (+8 for lanes 16-31).  Row reductions via shfl_xor within 16 lanes.
// ---------------------------------------------------------------------------
__global__ __launch_bounds__(256) void attn_kernel(
    const __bf16* __restrict__ qb,           // (B*T, 64)
    const __bf16* __restrict__ kb,           // (B*T, 64)
    const __bf16* __restrict__ vT,           // (B, 64, T)
    float* __restrict__ out)                 // (B, T, 64)
{
    __shared__ __bf16 plds_all[8][16 * 32];  // per-wave P tile (16x32) = 1KB

    const int wave = threadIdx.x >> 5;
    const int l    = threadIdx.x & 31;
    const int n    = l & 15;
    const int hi   = l >> 4;
    const int kofs = hi ? 8 : 0;

    const int    qt      = blockIdx.x * 8 + wave;   // 0..1023
    const int    bidx    = qt >> 8;                  // 256 tiles / batch
    const int    q0      = (qt & 255) * 16;          // first query row (in batch)
    const size_t rowbase = (size_t)bidx * TDIM;

    __bf16* plds = plds_all[wave];

    // Load Q as two A-fragments (features 0-31 / 32-63).
    v16bf qfrag[2];
    {
        const __bf16* qp = qb + (rowbase + q0 + n) * HDIM;
#pragma unroll
        for (int t = 0; t < 2; ++t)
#pragma unroll
            for (int j = 0; j < 8; ++j) {
                qfrag[t][j]     = qp[t * 32 + kofs + j];
                qfrag[t][8 + j] = qp[t * 32 + 16 + kofs + j];
            }
    }

    v8f o[4];
#pragma unroll
    for (int t = 0; t < 4; ++t) o[t] = (v8f)0.0f;
    float mrow[8], lrow[8];
#pragma unroll
    for (int r = 0; r < 8; ++r) { mrow[r] = -__builtin_inff(); lrow[r] = 0.0f; }

    const float scale = 0.125f;              // 1/sqrt(64)
    const int   kend  = q0 + 15;

    for (int kbase = 0; kbase <= kend; kbase += 32) {
        // ---- S = Q @ K^T for 32 keys (two 16-key N-tiles) ----
        v8f s[2];
        s[0] = (v8f)0.0f; s[1] = (v8f)0.0f;
#pragma unroll
        for (int st = 0; st < 2; ++st) {
            const __bf16* kp = kb + (rowbase + kbase + st * 16 + n) * HDIM;
#pragma unroll
            for (int t = 0; t < 2; ++t) {
                v16bf bf;
#pragma unroll
                for (int j = 0; j < 8; ++j) {
                    bf[j]     = kp[t * 32 + kofs + j];
                    bf[8 + j] = kp[t * 32 + 16 + kofs + j];
                }
                s[st] = __builtin_amdgcn_wmma_f32_16x16x32_bf16(
                    false, qfrag[t], false, bf, (short)0, s[st], false, false);
            }
        }

        // ---- scale + causal mask + online softmax ----
        const bool need_mask = (kbase + 31 > q0);
#pragma unroll
        for (int r = 0; r < 8; ++r) {
            const int qrow = q0 + r + 8 * hi;
            float v0 = s[0][r] * scale;
            float v1 = s[1][r] * scale;
            if (need_mask) {
                if (kbase + n      > qrow) v0 = -__builtin_inff();
                if (kbase + 16 + n > qrow) v1 = -__builtin_inff();
            }
            float mx = fmaxf(v0, v1);
            mx = fmaxf(mx, __shfl_xor(mx, 1, 32));
            mx = fmaxf(mx, __shfl_xor(mx, 2, 32));
            mx = fmaxf(mx, __shfl_xor(mx, 4, 32));
            mx = fmaxf(mx, __shfl_xor(mx, 8, 32));

            const float mnew  = fmaxf(mrow[r], mx);
            const float alpha = __expf(mrow[r] - mnew);
            mrow[r] = mnew;

            const float p0 = __expf(v0 - mnew);
            const float p1 = __expf(v1 - mnew);
            s[0][r] = p0;
            s[1][r] = p1;

            float rs = p0 + p1;
            rs += __shfl_xor(rs, 1, 32);
            rs += __shfl_xor(rs, 2, 32);
            rs += __shfl_xor(rs, 4, 32);
            rs += __shfl_xor(rs, 8, 32);
            lrow[r] = lrow[r] * alpha + rs;

#pragma unroll
            for (int t = 0; t < 4; ++t) o[t][r] *= alpha;
        }

        // ---- C-layout f32 P -> LDS -> A-layout bf16 P ----
#pragma unroll
        for (int st = 0; st < 2; ++st)
#pragma unroll
            for (int r = 0; r < 8; ++r)
                plds[(r + 8 * hi) * 32 + st * 16 + n] = (__bf16)s[st][r];

        v16bf pa;
        {
            const __bf16* pr = plds + n * 32;
#pragma unroll
            for (int j = 0; j < 8; ++j) {
                pa[j]     = pr[kofs + j];
                pa[8 + j] = pr[16 + kofs + j];
            }
        }

        // ---- O += P @ V  (V^T gives contiguous key chunks per lane) ----
#pragma unroll
        for (int t = 0; t < 4; ++t) {
            const __bf16* vp =
                vT + ((size_t)bidx * HDIM + t * 16 + n) * TDIM + kbase;
            v16bf bf;
#pragma unroll
            for (int j = 0; j < 8; ++j) {
                bf[j]     = vp[kofs + j];
                bf[8 + j] = vp[16 + kofs + j];
            }
            o[t] = __builtin_amdgcn_wmma_f32_16x16x32_bf16(
                false, pa, false, bf, (short)0, o[t], false, false);
        }
    }

    // ---- epilogue: divide by softmax denom, write fp32 output ----
#pragma unroll
    for (int t = 0; t < 4; ++t)
#pragma unroll
        for (int r = 0; r < 8; ++r) {
            const int qrow = q0 + r + 8 * hi;
            out[(rowbase + qrow) * HDIM + t * 16 + n] = o[t][r] / lrow[r];
        }
}

// ---------------------------------------------------------------------------
extern "C" void kernel_launch(void* const* d_in, const int* in_sizes, int n_in,
                              void* d_out, int out_size, void* d_ws, size_t ws_size,
                              hipStream_t stream) {
    const float* x  = (const float*)d_in[0];
    const float* Wq = (const float*)d_in[1];
    const float* bq = (const float*)d_in[2];
    const float* Wk = (const float*)d_in[3];
    const float* bk = (const float*)d_in[4];
    const float* Wv = (const float*)d_in[5];
    const float* bv = (const float*)d_in[6];
    float* out = (float*)d_out;

    // Workspace layout (bf16 halves):
    //   [0, 196608)                 packed W fragments (3*32*4*512)
    //   [196608, +1048576)          q bf16 (B*T*64)
    //   [.., +1048576)              k bf16
    //   [.., +1048576)              v^T bf16 (B,64,T)
    // Total = 6,684,672 bytes.
    __bf16* frags = (__bf16*)d_ws;
    __bf16* qbf   = frags + 196608;
    __bf16* kbf   = qbf + 1048576;
    __bf16* vTb   = kbf + 1048576;

    pack_weights_kernel<<<384, 32, 0, stream>>>(Wq, Wk, Wv, frags);
    qkv_kernel<<<128, 256, 0, stream>>>(x, bq, bk, bv, frags, qbf, kbf, vTb);
    attn_kernel<<<128, 256, 0, stream>>>(qbf, kbf, vTb, out);
}
